// JointERE_85048942395645
// MI455X (gfx1250) — matrix operand order repaired and checked by reading, not compiled, not run on hardware
//
#include <hip/hip_runtime.h>

#define Bn 32
#define Ln 1024
#define Dn 1024
#define Hn 16
#define DVn 64

typedef __attribute__((ext_vector_type(16))) __bf16 v16bf;
typedef __attribute__((ext_vector_type(8)))  __bf16 bf16x8;
typedef __attribute__((ext_vector_type(8)))  float  v8f;
typedef __attribute__((ext_vector_type(4)))  unsigned int v4u;
typedef __attribute__((ext_vector_type(8)))  int v8i;
typedef __attribute__((ext_vector_type(4)))  int v4i;

// Workspace layout (bytes)
#define OFF_Q      ((size_t)0)                       // fp32 [B][D]            128 KB
#define OFF_QB     ((size_t)131072)                  // fp32 [B][H]            2 KB
#define OFF_ABASE  ((size_t)133120)                  // fp32 [B][H]            2 KB
#define OFF_U      ((size_t)135168)                  // bf16 [B][H][D]         1 MB
#define OFF_S      ((size_t)1183744)                 // fp32 [B][H][L]         2 MB
#define OFF_WVT    ((size_t)3280896)                 // bf16 [N=D][K=D]        2 MB

__device__ __forceinline__ unsigned short f32_to_bf16_rne(float f) {
    union { float f; unsigned u; } x; x.f = f;
    unsigned r = x.u + 0x7FFFu + ((x.u >> 16) & 1u);
    return (unsigned short)(r >> 16);
}

__device__ __forceinline__ unsigned pack_bf16x2(float lo, float hi) {
    unsigned short a = __builtin_bit_cast(unsigned short, (__bf16)lo);
    unsigned short b = __builtin_bit_cast(unsigned short, (__bf16)hi);
    return (unsigned)a | ((unsigned)b << 16);
}

// ---------------- wvT[n][k] = bf16(wv[k][n]) (one-time, lands in L2) -----------
__global__ __launch_bounds__(256) void k_transpose_wv(const float* __restrict__ wv,
                                                      unsigned short* __restrict__ wvT) {
    __shared__ float tile[32][33];
    int tx = threadIdx.x, ty = threadIdx.y;
    int n0 = blockIdx.x * 32, k0 = blockIdx.y * 32;
#pragma unroll
    for (int j = 0; j < 4; ++j)
        tile[ty + j*8][tx] = wv[(size_t)(k0 + ty + j*8)*Dn + n0 + tx];
    __syncthreads();
#pragma unroll
    for (int j = 0; j < 4; ++j)
        wvT[(size_t)(n0 + ty + j*8)*Dn + k0 + tx] = f32_to_bf16_rne(tile[tx][ty + j*8]);
}

// ---------------- q[b,:] = enc[b, L-1, :] @ wq + bq ----------------------------
__global__ __launch_bounds__(256) void k_qproj(const float* __restrict__ enc,
                                               const float* __restrict__ wq,
                                               const float* __restrict__ bq,
                                               float* __restrict__ q) {
    int b = blockIdx.x, tid = threadIdx.x;
    __shared__ float row[Dn];
    const float* er = enc + ((size_t)b*Ln + (Ln - 1))*Dn;
    for (int i = tid; i < Dn; i += 256) row[i] = er[i];
    __syncthreads();
    float acc[4];
#pragma unroll
    for (int i = 0; i < 4; ++i) acc[i] = bq[tid + i*256];
    for (int d = 0; d < Dn; ++d) {
        float e = row[d];
        const float* wr = wq + (size_t)d*Dn + tid;
#pragma unroll
        for (int i = 0; i < 4; ++i) acc[i] += e * wr[i*256];
    }
#pragma unroll
    for (int i = 0; i < 4; ++i) q[(size_t)b*Dn + tid + i*256] = acc[i];
}

// ------- u[b,h,d] = sum_j wk[d, h*64+j] * q[b, h*64+j]   (folds K-GEMM away) ---
// ------- qb[b,h]  = sum_j bk[h*64+j]    * q[b, h*64+j] --------------------------
__global__ __launch_bounds__(256) void k_uproj(const float* __restrict__ wk,
                                               const float* __restrict__ bk,
                                               const float* __restrict__ q,
                                               unsigned short* __restrict__ u,
                                               float* __restrict__ qb) {
    __shared__ float qs[Bn*64];   // [b][j]
    __shared__ float wks[64*64];  // [dd][j]
    int tid = threadIdx.x;
    int d0 = blockIdx.x * 64, h = blockIdx.y;
    for (int i = tid; i < Bn*64; i += 256) {
        int bb = i >> 6, j = i & 63;
        qs[i] = q[(size_t)bb*Dn + h*64 + j];
    }
    for (int i = tid; i < 64*64; i += 256) {
        int dd = i >> 6, j = i & 63;
        wks[i] = wk[(size_t)(d0 + dd)*Dn + h*64 + j];
    }
    __syncthreads();
    for (int o = tid; o < 64*Bn; o += 256) {
        int dd = o >> 5, bb = o & 31;
        float acc = 0.f;
#pragma unroll 8
        for (int j = 0; j < 64; ++j) acc += wks[dd*64 + j] * qs[bb*64 + j];
        u[((size_t)bb*Hn + h)*Dn + d0 + dd] = f32_to_bf16_rne(acc);
    }
    if (blockIdx.x == 0 && tid < Bn) {
        float acc = 0.f;
        for (int j = 0; j < 64; ++j) acc += bk[h*64 + j] * qs[tid*64 + j];
        qb[tid*Hn + h] = acc;
    }
}

// ------- scores: s[b,h,l] = rate(l)/temp * (enc[b,l,:]·u[b,h,:] + qb[b,h]) -----
// WMMA bf16: M=16 seq rows per wave, N=16 heads, K=1024 in steps of 32.
__global__ __launch_bounds__(128) void k_scores(const float* __restrict__ enc,
                                                const unsigned short* __restrict__ u,
                                                const float* __restrict__ qb,
                                                float* __restrict__ s) {
    __shared__ unsigned short Atile[64*32];
    int tid = threadIdx.x;
    int b = blockIdx.y, lt = blockIdx.x;
    int wave = tid >> 5, lane = tid & 31;
    int half = lane >> 4, n = lane & 15;
    int row0 = lt*64 + wave*16;
    const unsigned short* uc = u + ((size_t)b*Hn + n)*Dn; // column = head n, K-contiguous
    const float* encB = enc + ((size_t)b*Ln + lt*64)*Dn;
    v8f c;
#pragma unroll
    for (int i = 0; i < 8; ++i) c[i] = 0.f;

    for (int k0 = 0; k0 < Dn; k0 += 32) {
        __syncthreads();
        // stage A tile (64 rows x 32 k) as bf16: packed pair converts, b32 LDS stores
#pragma unroll
        for (int i = 0; i < 8; ++i) {
            int p = tid + i*128;          // pair index, 1024 pairs
            int idx = p*2;
            int r = idx >> 5, kk = idx & 31;
            float2 v2 = *(const float2*)&encB[(size_t)r*Dn + k0 + kk];
            ((unsigned*)Atile)[p] = pack_bf16x2(v2.x, v2.y);
        }
        __syncthreads();
        v16bf a, bb;
        const bf16x8* ap0 = (const bf16x8*)&Atile[(wave*16 + n)*32 + half*16];
        const bf16x8* ap1 = (const bf16x8*)&Atile[(wave*16 + n)*32 + half*16 + 8];
        bf16x8 alo = *ap0, ahi = *ap1;
        const bf16x8* bp = (const bf16x8*)(uc + k0 + half*16);
        bf16x8 blo = bp[0], bhi = bp[1];
#pragma unroll
        for (int e = 0; e < 8; ++e) { a[e] = alo[e]; a[8+e] = ahi[e]; bb[e] = blo[e]; bb[8+e] = bhi[e]; }
        c = __builtin_amdgcn_wmma_f32_16x16x32_bf16(false, a, false, bb, (short)0, c, false, false);
    }
    float qbv = qb[b*Hn + n];
    const float invT = 1.0f / (32.0f + 1e-5f); // temperature = sqrt(1024)+1e-5
#pragma unroll
    for (int r = 0; r < 8; ++r) {
        int l = row0 + r + 8*half;
        float rate = (float)(Ln - l - 1) * (1.0f/(float)Ln) + 1.0f;
        s[((size_t)b*Hn + n)*Ln + l] = (c[r] + qbv) * rate * invT;
    }
}

// ---------------- logsumexp base per (b,h) -------------------------------------
__global__ __launch_bounds__(256) void k_lse(const float* __restrict__ s,
                                             float* __restrict__ abase) {
    __shared__ float red[256];
    int bh = blockIdx.x, tid = threadIdx.x;
    const float* sp = s + (size_t)bh*Ln;
    float m = -3.0e38f;
    for (int i = tid; i < Ln; i += 256) m = fmaxf(m, sp[i]);
    red[tid] = m; __syncthreads();
    for (int o = 128; o > 0; o >>= 1) { if (tid < o) red[tid] = fmaxf(red[tid], red[tid + o]); __syncthreads(); }
    m = red[0]; __syncthreads();
    float sum = 0.f;
    for (int i = tid; i < Ln; i += 256) sum += expf(sp[i] - m);
    red[tid] = sum; __syncthreads();
    for (int o = 128; o > 0; o >>= 1) { if (tid < o) red[tid] += red[tid + o]; __syncthreads(); }
    if (tid == 0) abase[bh] = m + logf(red[0]);
}

// ---- fused: V-GEMM (bf16 WMMA) + logattn*rate scale + residual + LayerNorm ----
// Block = 16 rows x 1024 cols (LN block-local). 8 waves x 8 N-tiles.
// A panel (16 x 512 fp32 per K-half) is DMA'd into LDS by the Tensor Data Mover:
// one wave issues TENSOR_LOAD_TO_LDS with a 2-D D#, waits TENSORcnt, then the
// whole K-half inner loop runs barrier-free out of LDS.
__global__ __launch_bounds__(256) void k_attnout(const float* __restrict__ enc,
                                                 const unsigned short* __restrict__ wvT,
                                                 const float* __restrict__ bv,
                                                 const float* __restrict__ s,
                                                 const float* __restrict__ abase,
                                                 const float* __restrict__ gamma,
                                                 const float* __restrict__ beta,
                                                 float* __restrict__ out) {
    __shared__ float Atile[16*512];   // 32 KB fp32 A panel (one K-half)
    __shared__ float attnT[Hn*16];
    __shared__ float lnsum[16], lnsq[16], lnmean[16], lnrstd[16];

    int tid = threadIdx.x;
    int b = blockIdx.y, l0 = blockIdx.x*16;
    int wave = tid >> 5, lane = tid & 31;
    int half = lane >> 4, n = lane & 15;
    int cb = wave*128;

    { int h = tid >> 4, li = tid & 15;
      attnT[h*16 + li] = s[((size_t)b*Hn + h)*Ln + l0 + li] - abase[b*Hn + h]; }

    const float* encB = enc + (size_t)b*Ln*Dn;

    size_t co[8];
#pragma unroll
    for (int t = 0; t < 8; ++t) co[t] = (size_t)(cb + t*16 + n)*Dn;

    v8f c[8];
#pragma unroll
    for (int t = 0; t < 8; ++t)
#pragma unroll
        for (int i = 0; i < 8; ++i) c[t][i] = 0.f;

    for (int kh = 0; kh < 2; ++kh) {
        __syncthreads();                      // previous half's LDS reads done
        if (wave == 0) {
            // Tensor DMA descriptor (D#), ISA ch.8: 2-D tile, 16 rows x 512 f32,
            // row stride 1024 elements, data_size=4B, type=2 ("image"), count=1.
            unsigned long long ga =
                (unsigned long long)(uintptr_t)(encB + (size_t)l0*Dn + kh*512) ;
            v4u g0;
            g0[0] = 1u;                                        // count=1, user mode
            g0[1] = (unsigned)(uintptr_t)(&Atile[0]);          // lds_addr (bytes)
            g0[2] = (unsigned)(ga & 0xFFFFFFFFu);              // global_addr[31:0]
            g0[3] = (unsigned)((ga >> 32) & 0x1FFFFFFu)        // global_addr[56:32]
                  | (2u << 30);                                // type=2
            v8i g1;
            g1[0] = (int)(2u << 16);                           // data_size=4B
            g1[1] = (int)(512u << 16);                         // tensor_dim0=512 (lo16 @63:48)
            g1[2] = (int)(16u  << 16);                         // tensor_dim1=16  (lo16 @95:80)
            g1[3] = (int)(512u << 16);                         // tile_dim0=512   (@127:112)
            g1[4] = 16;                                        // tile_dim1=16    (@143:128)
            g1[5] = 1024;                                      // tensor_dim0_stride
            g1[6] = 0;
            g1[7] = 0;
            v4i gz; gz[0] = 0; gz[1] = 0; gz[2] = 0; gz[3] = 0;
            v8i gz8; gz8[0] = 0; gz8[1] = 0; gz8[2] = 0; gz8[3] = 0;
            gz8[4] = 0; gz8[5] = 0; gz8[6] = 0; gz8[7] = 0;
            __builtin_amdgcn_tensor_load_to_lds(g0, g1, gz, gz, gz8, 0);
            __builtin_amdgcn_s_wait_tensorcnt(0);
        }
        __syncthreads();                      // panel visible to all waves

#pragma unroll 4
        for (int k0 = 0; k0 < 512; k0 += 32) {
            v16bf a;
            { const float* ar0 = &Atile[n*512 + k0 + half*16];
              const float* ar1 = &Atile[n*512 + k0 + half*16 + 8];
#pragma unroll
              for (int e = 0; e < 8; ++e) { a[e] = (__bf16)ar0[e]; a[8+e] = (__bf16)ar1[e]; }
            }
            if (k0 + 32 < 512)
                __builtin_prefetch((const void*)(wvT + co[(k0 >> 5) & 7] + kh*512 + k0 + 32), 0, 0);
#pragma unroll
            for (int t = 0; t < 8; ++t) {
                const bf16x8* bp = (const bf16x8*)(wvT + co[t] + kh*512 + k0 + half*16);
                bf16x8 blo = bp[0], bhi = bp[1];
                v16bf bbf;
#pragma unroll
                for (int e = 0; e < 8; ++e) { bbf[e] = blo[e]; bbf[8+e] = bhi[e]; }
                c[t] = __builtin_amdgcn_wmma_f32_16x16x32_bf16(false, a, false, bbf, (short)0, c[t], false, false);
            }
        }
    }

    // epilogue: out = logattn * rate * (v + bv) + residual, then LayerNorm
    float rate_r[8], sum_r[8], sq_r[8];
#pragma unroll
    for (int r = 0; r < 8; ++r) {
        int l = l0 + r + 8*half;
        rate_r[r] = (float)(Ln - l - 1) * (1.0f/(float)Ln) + 1.0f;
        sum_r[r] = 0.f; sq_r[r] = 0.f;
    }
#pragma unroll
    for (int t = 0; t < 8; ++t) {
        int col = cb + t*16 + n;
        int h = col >> 6;
        float bvv = bv[col];
#pragma unroll
        for (int r = 0; r < 8; ++r) {
            int lrow = r + 8*half;
            float v = attnT[h*16 + lrow] * rate_r[r] * (c[t][r] + bvv)
                    + encB[(size_t)(l0 + lrow)*Dn + col];
            c[t][r] = v;
            sum_r[r] += v;
            sq_r[r]  += v*v;
        }
    }
#pragma unroll
    for (int mk = 1; mk < 16; mk <<= 1) {
#pragma unroll
        for (int r = 0; r < 8; ++r) {
            sum_r[r] += __shfl_xor(sum_r[r], mk, 32);
            sq_r[r]  += __shfl_xor(sq_r[r],  mk, 32);
        }
    }
    __syncthreads();
    if (tid < 16) { lnsum[tid] = 0.f; lnsq[tid] = 0.f; }
    __syncthreads();
    if (n == 0) {
#pragma unroll
        for (int r = 0; r < 8; ++r) {
            int row = r + 8*half;
            atomicAdd(&lnsum[row], sum_r[r]);
            atomicAdd(&lnsq[row],  sq_r[r]);
        }
    }
    __syncthreads();
    if (tid < 16) {
        float mean = lnsum[tid] * (1.0f/(float)Dn);
        float var  = lnsq[tid] * (1.0f/(float)Dn) - mean*mean;
        lnmean[tid] = mean;
        lnrstd[tid] = rsqrtf(var + 1e-5f);
    }
    __syncthreads();
#pragma unroll
    for (int t = 0; t < 8; ++t) {
        int col = cb + t*16 + n;
        float g = gamma[col], be = beta[col];
#pragma unroll
        for (int r = 0; r < 8; ++r) {
            int lrow = r + 8*half;
            out[((size_t)b*Ln + l0 + lrow)*Dn + col] =
                (c[t][r] - lnmean[lrow]) * lnrstd[lrow] * g + be;
        }
    }
}

extern "C" void kernel_launch(void* const* d_in, const int* in_sizes, int n_in,
                              void* d_out, int out_size, void* d_ws, size_t ws_size,
                              hipStream_t stream) {
    (void)in_sizes; (void)n_in; (void)out_size; (void)ws_size;
    const float* enc = (const float*)d_in[0];
    const float* wq  = (const float*)d_in[1];
    const float* bq  = (const float*)d_in[2];
    const float* wk  = (const float*)d_in[3];
    const float* bk  = (const float*)d_in[4];
    const float* wv  = (const float*)d_in[5];
    const float* bv  = (const float*)d_in[6];
    const float* g   = (const float*)d_in[7];
    const float* be  = (const float*)d_in[8];
    float* out = (float*)d_out;

    char* ws = (char*)d_ws;
    float*          q     = (float*)(ws + OFF_Q);
    float*          qb    = (float*)(ws + OFF_QB);
    float*          abase = (float*)(ws + OFF_ABASE);
    unsigned short* u     = (unsigned short*)(ws + OFF_U);
    float*          s     = (float*)(ws + OFF_S);
    unsigned short* wvT   = (unsigned short*)(ws + OFF_WVT);

    k_transpose_wv<<<dim3(32, 32), dim3(32, 8), 0, stream>>>(wv, wvT);
    k_qproj      <<<dim3(32),      dim3(256),   0, stream>>>(enc, wq, bq, q);
    k_uproj      <<<dim3(16, 16),  dim3(256),   0, stream>>>(wk, bk, q, u, qb);
    k_scores     <<<dim3(16, 32),  dim3(128),   0, stream>>>(enc, u, qb, s);
    k_lse        <<<dim3(Bn*Hn),   dim3(256),   0, stream>>>(s, abase);
    k_attnout    <<<dim3(Ln/16, Bn), dim3(256), 0, stream>>>(enc, wvT, bv, s, abase, g, be, out);
}
